// Attention_38285338476631
// MI455X (gfx1250) — matrix operand order repaired and checked by reading, not compiled, not run on hardware
//
#include <hip/hip_runtime.h>
#include <hip/hip_bf16.h>
#include <stdint.h>

#define DIM   768
#define HEADS 12
#define DHEAD 64
#define BATCH 4
#define SEQ   1024

typedef __attribute__((ext_vector_type(16))) __bf16 v16bf;
typedef __attribute__((ext_vector_type(8)))  float  v8f;

static __device__ __forceinline__ unsigned short f2bf(float f) {
    unsigned int u = __builtin_bit_cast(unsigned int, f);
    u += 0x7FFFu + ((u >> 16) & 1u);           // round-to-nearest-even
    return (unsigned short)(u >> 16);
}

union Frag {
    v16bf v;
    uint4 q[2];
};

// A fragment (16x32 bf16): lane<16 holds row M=lane, K-chunks {k0..k0+7, k0+16..k0+23};
// lane>=16 holds same rows, K-chunks shifted by 8. Source: row-major [M,K] bf16.
static __device__ __forceinline__ v16bf load_frag_a(const unsigned short* p, int ld, int k0) {
    int lane = threadIdx.x & 31;
    int sel  = lane >> 4;
    const unsigned short* s = p + (size_t)(lane & 15) * ld + k0 + sel * 8;
    Frag f;
    f.q[0] = *(const uint4*)(s);
    f.q[1] = *(const uint4*)(s + 16);
    return f.v;
}

// B fragment (32x16 bf16, K x N): lane%16 = N, K-chunks {k0..k0+7, k0+16..k0+23}, lane-half shifts by 8.
// Source: row-major [N,K] bf16 (i.e. element B[k][n] = p[n*ld + k]).
static __device__ __forceinline__ v16bf load_frag_b(const unsigned short* p, int ld, int k0) {
    int lane = threadIdx.x & 31;
    int sel  = lane >> 4;
    const unsigned short* s = p + (size_t)(lane & 15) * ld + k0 + sel * 8;
    Frag f;
    f.q[0] = *(const uint4*)(s);
    f.q[1] = *(const uint4*)(s + 16);
    return f.v;
}

static __device__ __forceinline__ v8f wmma_bf16(v16bf a, v16bf b, v8f c) {
    return __builtin_amdgcn_wmma_f32_16x16x32_bf16(false, a, false, b, (short)0, c, false, false);
}

__global__ void cvt_bf16(const float* __restrict__ src, unsigned short* __restrict__ dst, int n) {
    int i = blockIdx.x * blockDim.x + threadIdx.x;
    if (i < n) dst[i] = f2bf(src[i]);
}

// One wave computes a 16x64 output tile of D = A @ W^T (A:[M,K] bf16, W:[N,K] bf16).
// mode 0: store bf16 as qh/kh [b,h,tok,d]   (head-split, row-major over d)
// mode 1: store bf16 as vT    [b,h,d,tok]   (head-split, transposed)
// mode 2: store f32 out[m, c] = acc + bias[c]
__global__ __launch_bounds__(256) void gemm16x64(
    const unsigned short* __restrict__ A, const unsigned short* __restrict__ W,
    int M, int K, int nt64, int mode,
    unsigned short* __restrict__ outb, float* __restrict__ outf,
    const float* __restrict__ bias)
{
    int wid   = (int)((blockIdx.x * blockDim.x + threadIdx.x) >> 5);
    int total = (M >> 4) * nt64;
    if (wid >= total) return;                    // uniform per wave: EXEC stays all-ones
    int mt = wid / nt64;
    int nt = wid - mt * nt64;
    int lane = threadIdx.x & 31;
    int sel = lane >> 4, ln = lane & 15;

    const unsigned short* Ar = A + (size_t)mt * 16 * K;
    const unsigned short* Wr = W + (size_t)nt * 64 * K;

    v8f acc[4];
#pragma unroll
    for (int j = 0; j < 4; ++j)
#pragma unroll
        for (int i = 0; i < 8; ++i) acc[j][i] = 0.0f;

    for (int k0 = 0; k0 < K; k0 += 32) {
        v16bf a = load_frag_a(Ar, K, k0);
        acc[0] = wmma_bf16(a, load_frag_b(Wr,                   K, k0), acc[0]);
        acc[1] = wmma_bf16(a, load_frag_b(Wr + (size_t)16 * K,  K, k0), acc[1]);
        acc[2] = wmma_bf16(a, load_frag_b(Wr + (size_t)32 * K,  K, k0), acc[2]);
        acc[3] = wmma_bf16(a, load_frag_b(Wr + (size_t)48 * K,  K, k0), acc[3]);
    }

#pragma unroll
    for (int j = 0; j < 4; ++j) {
#pragma unroll
        for (int i = 0; i < 8; ++i) {
            int row = mt * 16 + i + 8 * sel;     // C/D layout: VGPR i, lane-half selects +8
            int col = nt * 64 + j * 16 + ln;
            float val = acc[j][i];
            if (mode == 2) {
                outf[(size_t)row * (size_t)(nt64 * 64) + col] = val + bias[col];
            } else {
                int b    = row >> 10;            // / SEQ
                int ntok = row & (SEQ - 1);
                int d    = j * 16 + ln;          // nt indexes the head when nt64==HEADS
                size_t idx = (mode == 0)
                    ? ((((size_t)b * HEADS + nt) * SEQ + ntok) * DHEAD + d)
                    : ((((size_t)b * HEADS + nt) * DHEAD + d) * SEQ + ntok);
                outb[idx] = f2bf(val);
            }
        }
    }
}

// Flash attention with additive bias: out = softmax(bias + scale*q k^T) v.
// (log_softmax inside the reference's outer softmax cancels by shift invariance.)
// One wave per (b,h, 16-row tile); streams S in steps of 32 columns.
__global__ __launch_bounds__(128) void flash_attn(
    const unsigned short* __restrict__ qh, const unsigned short* __restrict__ kh,
    const unsigned short* __restrict__ vT, const float* __restrict__ bias,
    unsigned short* __restrict__ Y)
{
    __shared__ __align__(16) unsigned short stage[4][16 * 32];  // 1 KB per wave
    const int N = SEQ, D = DHEAD;
    int lane = threadIdx.x & 31;
    int wv   = threadIdx.x >> 5;
    int wid  = blockIdx.x * 4 + wv;
    int mt   = wid & 63;          // N/16 = 64 row tiles
    int bh   = wid >> 6;          // b*HEADS + h
    int sel = lane >> 4, ln = lane & 15;

    const unsigned short* qp = qh + ((size_t)bh * N + (size_t)mt * 16) * D;
    const unsigned short* kp = kh + (size_t)bh * N * D;
    const unsigned short* vp = vT + (size_t)bh * D * N;
    const float*          wp = bias + (size_t)bh * N * N + (size_t)mt * 16 * N;

    v16bf qa0 = load_frag_a(qp, D, 0);
    v16bf qa1 = load_frag_a(qp, D, 32);

    v8f O0, O1, O2, O3;
    float mrow[8], lrow[8];
#pragma unroll
    for (int i = 0; i < 8; ++i) {
        O0[i] = 0.f; O1[i] = 0.f; O2[i] = 0.f; O3[i] = 0.f;
        mrow[i] = -3.0e38f; lrow[i] = 0.f;
    }
    const float scale = 0.125f;   // d^-0.5, d = 64

    for (int s0 = 0; s0 < N; s0 += 32) {
        // Hide the HBM bias stream (the only 200MB-class traffic) behind WMMA work:
        // prefetch next step's 16x32 f32 bias tile while this step computes.
        if (s0 + 32 < N) {
#pragma unroll
            for (int i = 0; i < 8; ++i) {
                int row = i + 8 * sel;
                __builtin_prefetch(wp + (size_t)row * N + s0 + 32 + ln * 2, 0, 3);
            }
        }

        v8f S0, S1;
#pragma unroll
        for (int i = 0; i < 8; ++i) { S0[i] = 0.f; S1[i] = 0.f; }
        // scores: K-dim = 64 head channels, N-dim = 2 tiles of 16 sequence cols
        S0 = wmma_bf16(qa0, load_frag_b(kp + (size_t)s0 * D,        D, 0),  S0);
        S0 = wmma_bf16(qa1, load_frag_b(kp + (size_t)s0 * D,        D, 32), S0);
        S1 = wmma_bf16(qa0, load_frag_b(kp + (size_t)(s0 + 16) * D, D, 0),  S1);
        S1 = wmma_bf16(qa1, load_frag_b(kp + (size_t)(s0 + 16) * D, D, 32), S1);

#pragma unroll
        for (int i = 0; i < 8; ++i) {
            int row = i + 8 * sel;
            float z0 = wp[(size_t)row * N + s0 + ln]      + scale * S0[i];
            float z1 = wp[(size_t)row * N + s0 + 16 + ln] + scale * S1[i];
            // row max across the 16 lanes of this half-wave (masks <=8 stay in-half)
            float t = fmaxf(z0, z1);
            t = fmaxf(t, __shfl_xor(t, 1, 32));
            t = fmaxf(t, __shfl_xor(t, 2, 32));
            t = fmaxf(t, __shfl_xor(t, 4, 32));
            t = fmaxf(t, __shfl_xor(t, 8, 32));
            float nm    = fmaxf(mrow[i], t);
            float alpha = __expf(mrow[i] - nm);
            float p0 = __expf(z0 - nm);
            float p1 = __expf(z1 - nm);
            float rs = p0 + p1;
            rs += __shfl_xor(rs, 1, 32);
            rs += __shfl_xor(rs, 2, 32);
            rs += __shfl_xor(rs, 4, 32);
            rs += __shfl_xor(rs, 8, 32);
            lrow[i] = lrow[i] * alpha + rs;
            mrow[i] = nm;
            O0[i] *= alpha; O1[i] *= alpha; O2[i] *= alpha; O3[i] *= alpha;
            // C/D layout -> row-major [16][32] bf16 in LDS
            stage[wv][row * 32 + ln]      = f2bf(p0);
            stage[wv][row * 32 + 16 + ln] = f2bf(p1);
        }
        asm volatile("s_wait_dscnt 0" ::: "memory");
        v16bf pa = load_frag_a(&stage[wv][0], 32, 0);   // P as A fragment (16x32)
        // O += P @ V : B fragment from vT[d][s], 8 contiguous s at s0+8h and 8 at s0+16+8h per lane
        O0 = wmma_bf16(pa, load_frag_b(vp,                   N, s0), O0);
        O1 = wmma_bf16(pa, load_frag_b(vp + (size_t)16 * N,  N, s0), O1);
        O2 = wmma_bf16(pa, load_frag_b(vp + (size_t)32 * N,  N, s0), O2);
        O3 = wmma_bf16(pa, load_frag_b(vp + (size_t)48 * N,  N, s0), O3);
    }

    int b = bh / HEADS, h = bh - b * HEADS;
#pragma unroll
    for (int i = 0; i < 8; ++i) {
        float r  = 1.0f / lrow[i];
        int row  = i + 8 * sel;
        int tok  = mt * 16 + row;
        size_t base = ((size_t)(b * SEQ + tok)) * DIM + (size_t)h * DHEAD;
        Y[base + ln]      = f2bf(O0[i] * r);
        Y[base + 16 + ln] = f2bf(O1[i] * r);
        Y[base + 32 + ln] = f2bf(O2[i] * r);
        Y[base + 48 + ln] = f2bf(O3[i] * r);
    }
}

extern "C" void kernel_launch(void* const* d_in, const int* in_sizes, int n_in,
                              void* d_out, int out_size, void* d_ws, size_t ws_size,
                              hipStream_t stream)
{
    const float* query = (const float*)d_in[0];
    const float* attnw = (const float*)d_in[1];
    const float* Wq    = (const float*)d_in[2];
    const float* Wk    = (const float*)d_in[3];
    const float* Wv    = (const float*)d_in[4];
    const float* Wo    = (const float*)d_in[5];
    const float* bo    = (const float*)d_in[6];
    float* out = (float*)d_out;

    const size_t E_X = (size_t)BATCH * SEQ * DIM;   // 3,145,728
    const size_t E_W = (size_t)DIM * DIM;           //   589,824
    unsigned short* Xb  = (unsigned short*)d_ws;
    unsigned short* Wqb = Xb  + E_X;
    unsigned short* Wkb = Wqb + E_W;
    unsigned short* Wvb = Wkb + E_W;
    unsigned short* Wob = Wvb + E_W;
    unsigned short* Qh  = Wob + E_W;
    unsigned short* Kh  = Qh  + E_X;
    unsigned short* Vt  = Kh  + E_X;
    unsigned short* Yb  = Vt  + E_X;                // total ~36.2 MB

    cvt_bf16<<<(int)((E_X + 255) / 256), 256, 0, stream>>>(query, Xb,  (int)E_X);
    cvt_bf16<<<(int)((E_W + 255) / 256), 256, 0, stream>>>(Wq,    Wqb, (int)E_W);
    cvt_bf16<<<(int)((E_W + 255) / 256), 256, 0, stream>>>(Wk,    Wkb, (int)E_W);
    cvt_bf16<<<(int)((E_W + 255) / 256), 256, 0, stream>>>(Wv,    Wvb, (int)E_W);
    cvt_bf16<<<(int)((E_W + 255) / 256), 256, 0, stream>>>(Wo,    Wob, (int)E_W);

    const int M = BATCH * SEQ;                      // 4096
    const int nt64 = DIM / 64;                      // 12 (== HEADS)
    const int waves  = (M / 16) * nt64;             // 3072
    const int blocks = waves * 32 / 256;            // 384

    gemm16x64<<<blocks, 256, 0, stream>>>(Xb, Wqb, M, DIM, nt64, 0, Qh, nullptr, nullptr);
    gemm16x64<<<blocks, 256, 0, stream>>>(Xb, Wkb, M, DIM, nt64, 0, Kh, nullptr, nullptr);
    gemm16x64<<<blocks, 256, 0, stream>>>(Xb, Wvb, M, DIM, nt64, 1, Vt, nullptr, nullptr);

    flash_attn<<<(BATCH * HEADS * (SEQ / 16)) / 4, 128, 0, stream>>>(Qh, Kh, Vt, attnw, Yb);

    gemm16x64<<<blocks, 256, 0, stream>>>(Yb, Wob, M, DIM, nt64, 2, nullptr, out, bo);
}
